// Seq2SeqAttentionGRU_58394375356761
// MI455X (gfx1250) — matrix operand order repaired and checked by reading, not compiled, not run on hardware
//
#include <hip/hip_runtime.h>
#include <hip/hip_bf16.h>
#include <math.h>

// ---------------- problem constants ----------------
constexpr int kV  = 32000;
constexpr int kH  = 512;
constexpr int kH3 = 3 * kH;     // 1536
constexpr int kB  = 32;
constexpr int kS  = 128;
constexpr int kT  = 65;
constexpr int kTD = 64;         // decoder steps = T-1
constexpr int kBOS = 1;

typedef __attribute__((ext_vector_type(16))) _Float16 v16h;
typedef __attribute__((ext_vector_type(8)))  float    v8f;

// ---------------- WMMA GEMM: C[M,N] = A[M,K](f16) * W[N,K]^T(f16) + bias ----
// Register-blocked + software-pipelined: each wave computes a 16(M) x 64(N)
// slab of C (4 accumulators). K loop is double-buffered: while the 4 WMMAs of
// chunk k execute, the b128 loads of chunk k+32 are in flight in a separate
// register set, so the backend can overlap vmem with the matrix pipe instead
// of s_wait_loadcnt 0 on every fragment.
// grid = (N/256, M/16), block = 128 (4 waves along N).
// Fragment layouts used by this kernel (wave32):
//  A 16x32 f16 : lane l holds row m=l&15; k = (l>>4)*16 + [0..15]
//  B 32x16 f16 : lane l holds col n=l&15; k = (l>>4)*16 + [0..15]
//  C 16x16 f32 : lane l, vgpr r -> row (l>>4)*8 + r, col l&15
__global__ __launch_bounds__(128)
void wmma_gemm(const _Float16* __restrict__ A, int lda,
               const _Float16* __restrict__ W, int ldw,
               const float* __restrict__ bias,
               float* __restrict__ C, int ldc, int K)
{
    const int wave = threadIdx.x >> 5;
    const int lane = threadIdx.x & 31;
    const int hi   = lane >> 4;        // 0 or 1 (lane group)
    const int lo   = lane & 15;
    const int n0   = (blockIdx.x * 4 + wave) * 64;   // 4 N-tiles per wave
    const int m0   = blockIdx.y * 16;

    v8f acc[4];
#pragma unroll
    for (int j = 0; j < 4; ++j) {
        float b0 = bias ? bias[n0 + j * 16 + lo] : 0.0f;
#pragma unroll
        for (int r = 0; r < 8; ++r) acc[j][r] = b0;
    }

    const _Float16* Arow = A + (size_t)(m0 + lo) * lda + hi * 16;
    const _Float16* Wrow = W + (size_t)(n0 + lo) * ldw + hi * 16;
    const size_t wstep = (size_t)16 * ldw;

    v16h a0, a1, b0[4], b1[4];

    // prologue: chunk 0 into buffer set 0
#pragma unroll
    for (int i = 0; i < 8; ++i) { a0[i] = Arow[i]; a0[8 + i] = Arow[8 + i]; }
#pragma unroll
    for (int j = 0; j < 4; ++j) {
        const _Float16* Wp = Wrow + (size_t)j * wstep;
#pragma unroll
        for (int i = 0; i < 16; ++i) b0[j][i] = Wp[i];
    }

    for (int kb = 0; kb < K; kb += 64) {
        // load chunk kb+32 into set 1 while set 0 computes
        {
            const _Float16* Ap = Arow + kb + 32;
#pragma unroll
            for (int i = 0; i < 8; ++i) { a1[i] = Ap[i]; a1[8 + i] = Ap[8 + i]; }
#pragma unroll
            for (int j = 0; j < 4; ++j) {
                const _Float16* Wp = Wrow + (size_t)j * wstep + kb + 32;
                __builtin_prefetch(Wp + 32, 0, 3);
#pragma unroll
                for (int i = 0; i < 16; ++i) b1[j][i] = Wp[i];
            }
        }
#pragma unroll
        for (int j = 0; j < 4; ++j)
            acc[j] = __builtin_amdgcn_wmma_f32_16x16x32_f16(
                false, a0, false, b0[j], (short)0, acc[j], false, false);

        // load chunk kb+64 into set 0 while set 1 computes
        if (kb + 64 < K) {
            const _Float16* Ap = Arow + kb + 64;
#pragma unroll
            for (int i = 0; i < 8; ++i) { a0[i] = Ap[i]; a0[8 + i] = Ap[8 + i]; }
#pragma unroll
            for (int j = 0; j < 4; ++j) {
                const _Float16* Wp = Wrow + (size_t)j * wstep + kb + 64;
                __builtin_prefetch(Wp + 32, 0, 3);
#pragma unroll
                for (int i = 0; i < 16; ++i) b0[j][i] = Wp[i];
            }
        }
#pragma unroll
        for (int j = 0; j < 4; ++j)
            acc[j] = __builtin_amdgcn_wmma_f32_16x16x32_f16(
                false, a1, false, b1[j], (short)0, acc[j], false, false);
    }

#pragma unroll
    for (int j = 0; j < 4; ++j) {
        float* Cp = C + (size_t)(m0 + hi * 8) * ldc + n0 + j * 16 + lo;
#pragma unroll
        for (int r = 0; r < 8; ++r) Cp[(size_t)r * ldc] = acc[j][r];
    }
}

// ---------------- helpers ----------------
__global__ void f32_to_f16_k(const float* __restrict__ in,
                             _Float16* __restrict__ out, int n)
{
    int i = blockIdx.x * 256 + threadIdx.x;
    if (i < n) out[i] = (_Float16)in[i];
}

__global__ void zero_h_k(float* h, _Float16* h16, int n)
{
    int i = blockIdx.x * 256 + threadIdx.x;
    if (i < n) { h[i] = 0.0f; h16[i] = (_Float16)0.0f; }
}

__global__ void gather_enc_k(const int* __restrict__ ids,
                             const float* __restrict__ emb,
                             _Float16* __restrict__ out)
{
    int row = blockIdx.x;                  // b*kS + s
    int tok = ids[row];
    const float* e = emb + (size_t)tok * kH;
    _Float16* o = out + (size_t)row * kH;
    for (int j = threadIdx.x; j < kH; j += blockDim.x) o[j] = (_Float16)e[j];
}

__global__ void gather_dec_k(const int* __restrict__ tgt,
                             const float* __restrict__ emb,
                             _Float16* __restrict__ out)
{
    int row = blockIdx.x;                  // b*kTD + t
    int b = row / kTD, t = row % kTD;
    int tok = (t == 0) ? kBOS : tgt[b * kT + t];
    const float* e = emb + (size_t)tok * kH;
    _Float16* o = out + (size_t)row * kH;
    for (int j = threadIdx.x; j < kH; j += blockDim.x) o[j] = (_Float16)e[j];
}

// GRU elementwise fuse. gi includes b_ih (via GEMM bias); gh includes b_hh.
__global__ void gru_fuse_enc_k(const float* __restrict__ gi_all, int s,
                               const float* __restrict__ gh,
                               float* __restrict__ h, _Float16* __restrict__ h16,
                               float* __restrict__ keys, _Float16* __restrict__ keys16)
{
    int b = blockIdx.x, j = threadIdx.x;   // kH threads
    const float* gi = gi_all + (size_t)(b * kS + s) * kH3;
    const float* gr = gh + (size_t)b * kH3;
    float hv = h[b * kH + j];
    float r  = 1.0f / (1.0f + expf(-(gi[j]        + gr[j])));
    float z  = 1.0f / (1.0f + expf(-(gi[kH + j]   + gr[kH + j])));
    float nn = tanhf(gi[2 * kH + j] + r * gr[2 * kH + j]);
    float hn = (1.0f - z) * nn + z * hv;
    h[b * kH + j] = hn;
    h16[b * kH + j] = (_Float16)hn;
    size_t kidx = (size_t)(b * kS + s) * kH + j;
    keys[kidx] = hn;
    keys16[kidx] = (_Float16)hn;
}

__global__ void gru_fuse_dec_k(const float* __restrict__ gix_all, int t,
                               const float* __restrict__ gic,
                               const float* __restrict__ gh,
                               float* __restrict__ h, _Float16* __restrict__ h16)
{
    int b = blockIdx.x, j = threadIdx.x;   // kH threads
    const float* gx = gix_all + (size_t)(b * kTD + t) * kH3;
    const float* gc = gic + (size_t)b * kH3;
    const float* gr = gh + (size_t)b * kH3;
    float hv = h[b * kH + j];
    float ir = gx[j]          + gc[j];
    float iz = gx[kH + j]     + gc[kH + j];
    float in = gx[2 * kH + j] + gc[2 * kH + j];
    float r  = 1.0f / (1.0f + expf(-(ir + gr[j])));
    float z  = 1.0f / (1.0f + expf(-(iz + gr[kH + j])));
    float nn = tanhf(in + r * gr[2 * kH + j]);
    float hn = (1.0f - z) * nn + z * hv;
    h[b * kH + j] = hn;
    h16[b * kH + j] = (_Float16)hn;
}

// Additive attention for one decoder step: scores -> softmax -> context (f16).
__global__ __launch_bounds__(128)
void attn_step_k(const float* __restrict__ q,        // (B,H), includes ba
                 const float* __restrict__ ua_keys,  // (B*S,H), includes bu
                 const float* __restrict__ Va,       // (H)
                 const float* __restrict__ bv,       // (1)
                 const float* __restrict__ keys,     // (B*S,H)
                 _Float16* __restrict__ ctx16)       // (B,H)
{
    int b = blockIdx.x, tid = threadIdx.x;           // 128 threads = kS
    __shared__ float sc[kS];
    __shared__ float red[128];

    const float* qb = q + (size_t)b * kH;
    const float* uk = ua_keys + (size_t)(b * kS + tid) * kH;
    float acc = 0.0f;
    for (int hh = 0; hh < kH; ++hh) acc += Va[hh] * tanhf(qb[hh] + uk[hh]);
    sc[tid] = acc + bv[0];
    __syncthreads();

    red[tid] = sc[tid]; __syncthreads();
    for (int off = 64; off > 0; off >>= 1) {
        if (tid < off) red[tid] = fmaxf(red[tid], red[tid + off]);
        __syncthreads();
    }
    float m = red[0]; __syncthreads();

    float e = expf(sc[tid] - m);
    sc[tid] = e; red[tid] = e; __syncthreads();
    for (int off = 64; off > 0; off >>= 1) {
        if (tid < off) red[tid] += red[tid + off];
        __syncthreads();
    }
    float inv = 1.0f / red[0];
    __syncthreads();

    for (int hh = tid; hh < kH; hh += 128) {
        float c = 0.0f;
        for (int s2 = 0; s2 < kS; ++s2)
            c += sc[s2] * keys[(size_t)(b * kS + s2) * kH + hh];
        ctx16[b * kH + hh] = (_Float16)(c * inv);
    }
}

// log_softmax over V for one decoder step, writes final output rows.
__global__ __launch_bounds__(256)
void log_softmax_k(const float* __restrict__ logits,  // (B,V), includes bout
                   float* __restrict__ out, int t)
{
    int b = blockIdx.x, tid = threadIdx.x;
    __shared__ float red[256];
    const float* lr = logits + (size_t)b * kV;

    float m = -3.0e38f;
    for (int v = tid; v < kV; v += 256) m = fmaxf(m, lr[v]);
    red[tid] = m; __syncthreads();
    for (int off = 128; off > 0; off >>= 1) {
        if (tid < off) red[tid] = fmaxf(red[tid], red[tid + off]);
        __syncthreads();
    }
    m = red[0]; __syncthreads();

    float s = 0.0f;
    for (int v = tid; v < kV; v += 256) s += expf(lr[v] - m);
    red[tid] = s; __syncthreads();
    for (int off = 128; off > 0; off >>= 1) {
        if (tid < off) red[tid] += red[tid + off];
        __syncthreads();
    }
    float lse = m + logf(red[0]);

    float* orow = out + ((size_t)b * kTD + t) * kV;
    for (int v = tid; v < kV; v += 256) orow[v] = lr[v] - lse;
}

// ---------------- host side ----------------
static void gemm(hipStream_t stream,
                 const _Float16* A, int lda, const _Float16* W, int ldw,
                 const float* bias, float* C, int ldc, int M, int N, int K)
{
    dim3 g(N / 256, M / 16);   // wave = 16x64 of C; block = 16x256
    wmma_gemm<<<g, 128, 0, stream>>>(A, lda, W, ldw, bias, C, ldc, K);
}

static void to_f16(hipStream_t stream, const float* in, _Float16* out, int n)
{
    f32_to_f16_k<<<(n + 255) / 256, 256, 0, stream>>>(in, out, n);
}

extern "C" void kernel_launch(void* const* d_in, const int* in_sizes, int n_in,
                              void* d_out, int out_size, void* d_ws, size_t ws_size,
                              hipStream_t stream)
{
    (void)in_sizes; (void)n_in; (void)out_size; (void)ws_size;

    const int*   input_ids  = (const int*)  d_in[0];
    const int*   target_ids = (const int*)  d_in[1];
    const float* emb        = (const float*)d_in[2];
    const float* enc_W_ih   = (const float*)d_in[3];
    const float* enc_W_hh   = (const float*)d_in[4];
    const float* enc_b_ih   = (const float*)d_in[5];
    const float* enc_b_hh   = (const float*)d_in[6];
    const float* Wa         = (const float*)d_in[7];
    const float* ba         = (const float*)d_in[8];
    const float* Ua         = (const float*)d_in[9];
    const float* bu         = (const float*)d_in[10];
    const float* Va         = (const float*)d_in[11];
    const float* bv         = (const float*)d_in[12];
    const float* dec_W_ih   = (const float*)d_in[13];
    const float* dec_W_hh   = (const float*)d_in[14];
    const float* dec_b_ih   = (const float*)d_in[15];
    const float* dec_b_hh   = (const float*)d_in[16];
    const float* Wout       = (const float*)d_in[17];
    const float* bout       = (const float*)d_in[18];
    float* out = (float*)d_out;

    // bump allocator over workspace
    char* p = (char*)d_ws;
    auto alloc = [&](size_t bytes) -> void* {
        void* r = (void*)p;
        p += (bytes + 255) & ~(size_t)255;
        return r;
    };

    _Float16* encWih16 = (_Float16*)alloc((size_t)kH3 * kH * 2);
    _Float16* encWhh16 = (_Float16*)alloc((size_t)kH3 * kH * 2);
    _Float16* Ua16     = (_Float16*)alloc((size_t)kH  * kH * 2);
    _Float16* Wa16     = (_Float16*)alloc((size_t)kH  * kH * 2);
    _Float16* decWih16 = (_Float16*)alloc((size_t)kH3 * 2 * kH * 2);
    _Float16* decWhh16 = (_Float16*)alloc((size_t)kH3 * kH * 2);
    _Float16* Wout16   = (_Float16*)alloc((size_t)kV  * kH * 2);
    _Float16* encx16   = (_Float16*)alloc((size_t)kB * kS  * kH * 2);
    _Float16* decx16   = (_Float16*)alloc((size_t)kB * kTD * kH * 2);
    float*    GI_enc   = (float*)   alloc((size_t)kB * kS  * kH3 * 4);
    float*    GIx_dec  = (float*)   alloc((size_t)kB * kTD * kH3 * 4);
    float*    keys     = (float*)   alloc((size_t)kB * kS * kH * 4);
    _Float16* keys16   = (_Float16*)alloc((size_t)kB * kS * kH * 2);
    float*    ua_keys  = (float*)   alloc((size_t)kB * kS * kH * 4);
    float*    h        = (float*)   alloc((size_t)kB * kH * 4);
    _Float16* h16      = (_Float16*)alloc((size_t)kB * kH * 2);
    float*    q        = (float*)   alloc((size_t)kB * kH * 4);
    _Float16* ctx16    = (_Float16*)alloc((size_t)kB * kH * 2);
    float*    gh       = (float*)   alloc((size_t)kB * kH3 * 4);
    float*    gic      = (float*)   alloc((size_t)kB * kH3 * 4);
    float*    logits   = (float*)   alloc((size_t)kB * kV * 4);

    // 1) weight conversion f32 -> f16 (once per launch; deterministic)
    to_f16(stream, enc_W_ih, encWih16, kH3 * kH);
    to_f16(stream, enc_W_hh, encWhh16, kH3 * kH);
    to_f16(stream, Ua, Ua16, kH * kH);
    to_f16(stream, Wa, Wa16, kH * kH);
    to_f16(stream, dec_W_ih, decWih16, kH3 * 2 * kH);
    to_f16(stream, dec_W_hh, decWhh16, kH3 * kH);
    to_f16(stream, Wout, Wout16, kV * kH);

    // 2) embedding gathers (direct f16 output)
    gather_enc_k<<<kB * kS, 128, 0, stream>>>(input_ids, emb, encx16);
    gather_dec_k<<<kB * kTD, 128, 0, stream>>>(target_ids, emb, decx16);

    // 3) batched input projections (the GRU parts independent of h)
    //    GI_enc = enc_x @ enc_W_ih^T + b_ih       (4096 x 1536, K=512)
    gemm(stream, encx16, kH, encWih16, kH, enc_b_ih, GI_enc, kH3,
         kB * kS, kH3, kH);
    //    GIx_dec = dec_x @ dec_W_ih[:, :H]^T + b_ih  (2048 x 1536, K=512)
    gemm(stream, decx16, kH, decWih16, 2 * kH, dec_b_ih, GIx_dec, kH3,
         kB * kTD, kH3, kH);

    // 4) encoder recurrence
    zero_h_k<<<(kB * kH + 255) / 256, 256, 0, stream>>>(h, h16, kB * kH);
    for (int s = 0; s < kS; ++s) {
        // gh = h @ enc_W_hh^T + b_hh   (32 x 1536, K=512)
        gemm(stream, h16, kH, encWhh16, kH, enc_b_hh, gh, kH3, kB, kH3, kH);
        gru_fuse_enc_k<<<kB, kH, 0, stream>>>(GI_enc, s, gh, h, h16, keys, keys16);
    }

    // 5) ua_keys = keys @ Ua^T + bu   (4096 x 512, K=512)
    gemm(stream, keys16, kH, Ua16, kH, bu, ua_keys, kH, kB * kS, kH, kH);

    // 6) decoder recurrence (h carries over from encoder final state)
    for (int t = 0; t < kTD; ++t) {
        // q = h @ Wa^T + ba            (32 x 512, K=512)
        gemm(stream, h16, kH, Wa16, kH, ba, q, kH, kB, kH, kH);
        attn_step_k<<<kB, 128, 0, stream>>>(q, ua_keys, Va, bv, keys, ctx16);
        // gic = ctx @ dec_W_ih[:, H:]^T  (32 x 1536, K=512, no bias)
        gemm(stream, ctx16, kH, decWih16 + kH, 2 * kH, nullptr, gic, kH3,
             kB, kH3, kH);
        // gh = h @ dec_W_hh^T + b_hh   (32 x 1536, K=512)
        gemm(stream, h16, kH, decWhh16, kH, dec_b_hh, gh, kH3, kB, kH3, kH);
        gru_fuse_dec_k<<<kB, kH, 0, stream>>>(GIx_dec, t, gic, gh, h, h16);
        // logits = h @ Wout^T + bout   (32 x 32000, K=512) -- dominant GEMM
        gemm(stream, h16, kH, Wout16, kH, bout, logits, kV, kB, kV, kH);
        log_softmax_k<<<kB, 256, 0, stream>>>(logits, out, t);
    }
}
